// Head_A2B_Lite_70952859730361
// MI455X (gfx1250) — matrix operand order repaired and checked by reading, not compiled, not run on hardware
//
#include <hip/hip_runtime.h>

#define NB 4
#define SA_ 2048
#define SB_ 2048
#define DD 1024   // a_n_embd (output dim / attention dim)
#define EE 1024   // b_n_embd (input dim of Linear)

typedef __bf16 bf16_t;
typedef __attribute__((ext_vector_type(4)))  float  v4f;
typedef __attribute__((ext_vector_type(8)))  float  v8f;
typedef __attribute__((ext_vector_type(8)))  bf16_t v8bf;
typedef __attribute__((ext_vector_type(16))) bf16_t v16bf;

static __device__ __forceinline__ v16bf cat8(v8bf lo, v8bf hi) {
  return __builtin_shufflevector(lo, hi, 0,1,2,3,4,5,6,7,8,9,10,11,12,13,14,15);
}

// A-fragment (16x32 bf16): lane (l%16)=row M, K chunks at half*8 and 16+half*8
static __device__ __forceinline__ v16bf ldA(const bf16_t* __restrict__ rowp, int k0, int half) {
  v8bf lo = *(const v8bf*)(rowp + k0 + half * 8);
  v8bf hi = *(const v8bf*)(rowp + k0 + 16 + half * 8);
  return cat8(lo, hi);
}
// B-fragment (32x16 bf16): lane (l%16)=col N, K chunks at half*8 and 16+half*8
static __device__ __forceinline__ v16bf ldB(const bf16_t* __restrict__ rowp, int k0, int half) {
  v8bf lo = *(const v8bf*)(rowp + k0 + half * 8);
  v8bf hi = *(const v8bf*)(rowp + k0 + 16 + half * 8);
  return cat8(lo, hi);
}

static __device__ __forceinline__ v8f wmma_bf16(v16bf a, v16bf b, v8f c) {
  return __builtin_amdgcn_wmma_f32_16x16x32_bf16(false, a, false, b, (short)0, c, false, false);
}

static __device__ __forceinline__ v8bf cvt8(const float* __restrict__ p) {
  v4f x0 = *(const v4f*)p;
  v4f x1 = *(const v4f*)(p + 4);
  v8bf r;
#pragma unroll
  for (int i = 0; i < 4; ++i) { r[i] = (bf16_t)x0[i]; r[i + 4] = (bf16_t)x1[i]; }
  return r;
}

// ---------------------------------------------------------------------------
// Kernel 1: a (f32, [B,SA,D]) -> a_bf16 [B,SA,D] and aT_bf16 [B,D,SA]
// ---------------------------------------------------------------------------
__global__ __launch_bounds__(256) void k_convert_a(const float* __restrict__ a,
                                                   bf16_t* __restrict__ a_bf,
                                                   bf16_t* __restrict__ aT_bf) {
  __shared__ bf16_t tile[32][33];
  const int bb = blockIdx.z;
  const int t0 = blockIdx.x * 32, d0 = blockIdx.y * 32;
#pragma unroll
  for (int k = 0; k < 4; ++k) {
    int t = t0 + threadIdx.y + k * 8;
    int d = d0 + threadIdx.x;
    float v = a[((size_t)bb * SA_ + t) * DD + d];
    bf16_t h = (bf16_t)v;
    a_bf[((size_t)bb * SA_ + t) * DD + d] = h;
    tile[threadIdx.y + k * 8][threadIdx.x] = h;
  }
  __syncthreads();
#pragma unroll
  for (int k = 0; k < 4; ++k) {
    int d = d0 + threadIdx.y + k * 8;
    int t = t0 + threadIdx.x;
    aT_bf[((size_t)bb * DD + d) * SA_ + t] = tile[threadIdx.x][threadIdx.y + k * 8];
  }
}

// ---------------------------------------------------------------------------
// Kernel 2: q = b @ Wq^T   (NT GEMM, f32 in -> bf16 frags inline -> bf16 out)
// grid (B*SB/16, D/128), block 128 (4 waves); wave w -> 2 N tiles, shared A cvt
// ---------------------------------------------------------------------------
__global__ __launch_bounds__(128) void k_qgemm(const float* __restrict__ bmat,
                                               const float* __restrict__ wq,
                                               bf16_t* __restrict__ qout) {
  const int lane = threadIdx.x & 31, w = threadIdx.x >> 5;
  const int lm = lane & 15, half = lane >> 4;
  const int s0 = blockIdx.x * 16;
  const int n0 = (blockIdx.y * 8 + w * 2) * 16;
  const float* arow  = bmat + (size_t)(s0 + lm) * EE;  // A: rows of b
  const float* brow0 = wq + (size_t)(n0 + lm) * EE;    // B: rows of Wq
  const float* brow1 = brow0 + (size_t)16 * EE;
  v8f c0 = {}, c1 = {};
#pragma unroll 2
  for (int e0 = 0; e0 < EE; e0 += 32) {
    v16bf af = cat8(cvt8(arow + e0 + half * 8), cvt8(arow + e0 + 16 + half * 8));
    v16bf b0 = cat8(cvt8(brow0 + e0 + half * 8), cvt8(brow0 + e0 + 16 + half * 8));
    v16bf b1 = cat8(cvt8(brow1 + e0 + half * 8), cvt8(brow1 + e0 + 16 + half * 8));
    c0 = wmma_bf16(af, b0, c0);
    c1 = wmma_bf16(af, b1, c1);
  }
#pragma unroll
  for (int v = 0; v < 8; ++v) {
    int m = v + half * 8;
    qout[(size_t)(s0 + m) * DD + n0 + lm]      = (bf16_t)c0[v];
    qout[(size_t)(s0 + m) * DD + n0 + 16 + lm] = (bf16_t)c1[v];
  }
}

// ---------------------------------------------------------------------------
// Kernel 3: fused attention. 1 block = 16 q rows; full-row softmax in LDS.
// ---------------------------------------------------------------------------
#define SSTR 2056  // 2048 + 8 bf16 pad (16B) -> conflict-free ds_load_b128

__global__ __launch_bounds__(128) void k_attn(const bf16_t* __restrict__ Q,
                                              const bf16_t* __restrict__ A,
                                              const bf16_t* __restrict__ AT,
                                              float* __restrict__ out) {
  extern __shared__ char smem[];
  bf16_t* sS = (bf16_t*)smem;                          // scores bf16 [16][SSTR]
  bf16_t* sP = (bf16_t*)(smem + 16 * SSTR * 2);        // exp(P)  bf16 [16][SSTR]
  float* red  = (float*)(smem + 2 * 16 * SSTR * 2);    // 128 partials
  float* rmax = red + 128;                             // 16
  float* rsum = rmax + 16;                             // 16

  const int tid = threadIdx.x;
  const int lane = tid & 31, w = tid >> 5;
  const int lm = lane & 15, half = lane >> 4;
  const int bb = blockIdx.x >> 7;            // SB/16 == 128 tiles per batch
  const int s0 = (blockIdx.x & 127) * 16;
  const float scale = 0.03125f;              // 1024^-0.5

  const bf16_t* qrow = Q + (size_t)(bb * SB_ + s0 + lm) * DD;

  // ---- Phase A: scores = (q . a) * scale, 4 KV tiles per pass ----
  for (int p = 0; p < 8; ++p) {
    const int tt = (w * 32 + 4 * p) * 16;
    const bf16_t* ar0 = A + (size_t)(bb * SA_ + tt + lm) * DD;
    const bf16_t* ar1 = ar0 + (size_t)16 * DD;
    const bf16_t* ar2 = ar0 + (size_t)32 * DD;
    const bf16_t* ar3 = ar0 + (size_t)48 * DD;
    v8f c0 = {}, c1 = {}, c2 = {}, c3 = {};
#pragma unroll 2
    for (int d0 = 0; d0 < DD; d0 += 32) {
      v16bf af = ldA(qrow, d0, half);
      v16bf b0 = ldB(ar0, d0, half);
      v16bf b1 = ldB(ar1, d0, half);
      v16bf b2 = ldB(ar2, d0, half);
      v16bf b3 = ldB(ar3, d0, half);
      c0 = wmma_bf16(af, b0, c0);
      c1 = wmma_bf16(af, b1, c1);
      c2 = wmma_bf16(af, b2, c2);
      c3 = wmma_bf16(af, b3, c3);
    }
#pragma unroll
    for (int v = 0; v < 8; ++v) {
      int m = v + half * 8;
      sS[m * SSTR + tt + lm]      = (bf16_t)(c0[v] * scale);
      sS[m * SSTR + tt + 16 + lm] = (bf16_t)(c1[v] * scale);
      sS[m * SSTR + tt + 32 + lm] = (bf16_t)(c2[v] * scale);
      sS[m * SSTR + tt + 48 + lm] = (bf16_t)(c3[v] * scale);
    }
  }
  __syncthreads();

  // ---- Phase B: softmax over full SA row ----
  {
    const int r = tid >> 3, j = tid & 7;
    const bf16_t* srow = sS + r * SSTR;
    float mloc = -3.0e38f;
    for (int c = j * 256; c < j * 256 + 256; ++c) mloc = fmaxf(mloc, (float)srow[c]);
    red[r * 8 + j] = mloc;
    __syncthreads();
    if (j == 0) {
      float m = red[r * 8];
#pragma unroll
      for (int k = 1; k < 8; ++k) m = fmaxf(m, red[r * 8 + k]);
      rmax[r] = m;
    }
    __syncthreads();
    const float m = rmax[r];
    float sloc = 0.0f;
    bf16_t* prow = sP + r * SSTR;
    for (int c = j * 256; c < j * 256 + 256; ++c) {
      float e = __expf((float)srow[c] - m);
      prow[c] = (bf16_t)e;
      sloc += e;
    }
    red[r * 8 + j] = sloc;
    __syncthreads();
    if (j == 0) {
      float s = 0.0f;
#pragma unroll
      for (int k = 0; k < 8; ++k) s += red[r * 8 + k];
      rsum[r] = s;
    }
    __syncthreads();
  }

  // ---- Phase C: out = (P @ a) / rowsum, 4 d tiles per pass ----
  const bf16_t* prow = sP + lm * SSTR;
  for (int p = 0; p < 4; ++p) {
    const int d0 = (w * 16 + 4 * p) * 16;
    const bf16_t* br0 = AT + (size_t)(bb * DD + d0 + lm) * SA_;
    const bf16_t* br1 = br0 + (size_t)16 * SA_;
    const bf16_t* br2 = br0 + (size_t)32 * SA_;
    const bf16_t* br3 = br0 + (size_t)48 * SA_;
    v8f c0 = {}, c1 = {}, c2 = {}, c3 = {};
#pragma unroll 2
    for (int t0 = 0; t0 < SA_; t0 += 32) {
      v16bf af = ldA(prow, t0, half);     // ds_load_b128 from LDS
      v16bf b0 = ldB(br0, t0, half);
      v16bf b1 = ldB(br1, t0, half);
      v16bf b2 = ldB(br2, t0, half);
      v16bf b3 = ldB(br3, t0, half);
      c0 = wmma_bf16(af, b0, c0);
      c1 = wmma_bf16(af, b1, c1);
      c2 = wmma_bf16(af, b2, c2);
      c3 = wmma_bf16(af, b3, c3);
    }
#pragma unroll
    for (int v = 0; v < 8; ++v) {
      int m = v + half * 8;
      float inv = 1.0f / rsum[m];
      size_t o = (size_t)(bb * SB_ + s0 + m) * DD + d0 + lm;
      out[o]      = c0[v] * inv;
      out[o + 16] = c1[v] * inv;
      out[o + 32] = c2[v] * inv;
      out[o + 48] = c3[v] * inv;
    }
  }
}

// ---------------------------------------------------------------------------
extern "C" void kernel_launch(void* const* d_in, const int* in_sizes, int n_in,
                              void* d_out, int out_size, void* d_ws, size_t ws_size,
                              hipStream_t stream) {
  (void)in_sizes; (void)n_in; (void)out_size; (void)ws_size;
  const float* a  = (const float*)d_in[0];
  const float* b  = (const float*)d_in[1];
  const float* wq = (const float*)d_in[2];
  float* out = (float*)d_out;

  char* ws = (char*)d_ws;
  bf16_t* a_bf  = (bf16_t*)ws;                                 // 16 MB
  bf16_t* aT_bf = (bf16_t*)(ws + (size_t)16 * 1024 * 1024);    // 16 MB
  bf16_t* q_bf  = (bf16_t*)(ws + (size_t)32 * 1024 * 1024);    // 16 MB

  dim3 g1(SA_ / 32, DD / 32, NB), b1(32, 8);
  k_convert_a<<<g1, b1, 0, stream>>>(a, a_bf, aT_bf);

  dim3 g2(NB * SB_ / 16, DD / 128), b2(128);
  k_qgemm<<<g2, b2, 0, stream>>>(b, wq, q_bf);

  dim3 g3(NB * SB_ / 16), b3(128);
  size_t shm = (size_t)2 * 16 * SSTR * 2 + 128 * 4 + 16 * 4 + 16 * 4;
  k_attn<<<g3, b3, shm, stream>>>(q_bf, a_bf, aT_bf, out);
}
